// TransformerDecoderLayer_88072599372039
// MI455X (gfx1250) — matrix operand order repaired and checked by reading, not compiled, not run on hardware
//
#include <hip/hip_runtime.h>
#include <hip/hip_bf16.h>

typedef __attribute__((ext_vector_type(16))) __bf16         v16bf;
typedef __attribute__((ext_vector_type(16))) unsigned short v16us;
typedef __attribute__((ext_vector_type(8)))  float          v8f;
typedef unsigned short ushort_t;

#define B_   2
#define S_   2048
#define D_   1024
#define H_   16
#define DK_  64
#define DFF_ 4096

// fp32 -> bf16, round-to-nearest-even
__device__ __forceinline__ unsigned short f2bf(float f) {
  unsigned u = __float_as_uint(f);
  u += 0x7FFFu + ((u >> 16) & 1u);
  return (unsigned short)(u >> 16);
}
__device__ __forceinline__ unsigned pack2(float lo, float hi) {
  return (unsigned)f2bf(lo) | ((unsigned)f2bf(hi) << 16);
}

// CDNA5 WMMA 16-bit A-matrix (16x32): element i -> contraction index
// (one contiguous 16-element run, same order as kB)
__device__ __forceinline__ int kA(int i, int lane) {
  return i + ((lane >> 4) << 4);
}
// CDNA5 WMMA 16-bit B-matrix (32x16): element i -> contraction index
// (one contiguous 16-element run)
__device__ __forceinline__ int kB(int i, int lane) {
  return i + ((lane >> 4) << 4);
}

__device__ __forceinline__ v8f wmma_bf16(v16us a, v16us b, v8f c) {
  return __builtin_amdgcn_wmma_f32_16x16x32_bf16(
      false, __builtin_bit_cast(v16bf, a),
      false, __builtin_bit_cast(v16bf, b),
      (short)0, c, false, false);
}

// v_permlane16_b32 cross-lane ops within each 16-lane row (pure VALU)
__device__ __forceinline__ float plane16(float x, int lo, int hi) {
  int v = __float_as_int(x);
  return __int_as_float(__builtin_amdgcn_permlane16(v, v, lo, hi, false, false));
}
#define X1_LO 0x67452301
#define X1_HI 0xefcdab89
#define X2_LO 0x54761032
#define X2_HI 0xdcfeba98
#define X4_LO 0x32107654
#define X4_HI 0xba98fedc
#define X8_LO 0xfedcba98
#define X8_HI 0x76543210

__device__ __forceinline__ float rowmax16(float x) {
  x = fmaxf(x, plane16(x, X1_LO, X1_HI));
  x = fmaxf(x, plane16(x, X2_LO, X2_HI));
  x = fmaxf(x, plane16(x, X4_LO, X4_HI));
  x = fmaxf(x, plane16(x, X8_LO, X8_HI));
  return x;
}

// ---------------------------------------------------------------------------
// One-time fp32 -> bf16 cast (vectorized, 8 elements/thread)
// ---------------------------------------------------------------------------
__global__ __launch_bounds__(256) void cast_f32_bf16(
    const float* __restrict__ src, ushort_t* __restrict__ dst, int n8) {
  int i = blockIdx.x * 256 + threadIdx.x;
  if (i >= n8) return;
  const float4* s = (const float4*)src;
  float4 a = s[2 * i], b = s[2 * i + 1];
  uint4 o;
  o.x = pack2(a.x, a.y);
  o.y = pack2(a.z, a.w);
  o.z = pack2(b.x, b.y);
  o.w = pack2(b.z, b.w);
  ((uint4*)dst)[i] = o;
}

// ---------------------------------------------------------------------------
// One-time transpose + cast: src f32 [K][N] -> dst bf16 [N][K]
// 64x64 tiles through LDS; both global phases coalesced.
// ---------------------------------------------------------------------------
__global__ __launch_bounds__(256) void cast_transpose_bf16(
    const float* __restrict__ src, ushort_t* __restrict__ dst, int K, int N) {
  __shared__ __align__(16) ushort_t t[64][72];
  const int tid = threadIdx.x;
  const int k0 = blockIdx.y * 64, n0 = blockIdx.x * 64;
#pragma unroll
  for (int i = tid; i < 4096; i += 256) {
    int r = i >> 6, c = i & 63;  // r: k-local, c: n-local
    t[c][r] = f2bf(src[(size_t)(k0 + r) * N + n0 + c]);
  }
  __syncthreads();
#pragma unroll
  for (int i = tid; i < 512; i += 256) {
    int r = i >> 3, c8 = (i & 7) * 8;  // r: n-local
    *(uint4*)(dst + (size_t)(n0 + r) * K + k0 + c8) = *(const uint4*)&t[r][c8];
  }
}

// ---------------------------------------------------------------------------
// Tiled GEMM, all-bf16 operands: out = A[M,K] @ Wt[N,K]^T + bias
// Block tile 128x128, BK=32, 256 threads = 8 waves, wave tile 32x64 (8 WMMAs).
// Both tiles stored K-contiguous in LDS -> all frag gathers are b128 loads.
// LDS double-buffered: one barrier/iteration, next tile loaded into registers
// while WMMAs run.
// MODE 0: f32 out                MODE 1: ReLU, bf16 out
// MODE 2: dual head-layout write (f32 out0 + bf16 out1, both [B,H,S,DK])
// MODE 3: head-layout bf16 out, pre-scaled by 1/sqrt(DK)   (Q)
// MODE 4: f32 [B,H,S,DK] out0 + bf16 transposed [B,H,DK,S] out1  (V)
// ---------------------------------------------------------------------------
template <int MODE>
__global__ __launch_bounds__(256) void gemm_bf16_wmma(
    const ushort_t* __restrict__ A, const ushort_t* __restrict__ Wt,
    const float* __restrict__ bias, void* __restrict__ out0,
    void* __restrict__ out1, int M, int N, int K) {
  __shared__ __align__(16) ushort_t As[2][128][40];  // 80B row stride
  __shared__ __align__(16) ushort_t Bs[2][128][40];
  const int tid  = threadIdx.x;
  const int lane = tid & 31;
  const int w    = tid >> 5;
  const int wm   = w & 3, wn = w >> 2;     // 4 x 2 wave grid
  const int bM   = blockIdx.y * 128, bN = blockIdx.x * 128;
  const int nl   = lane & 15;
  const int cr   = tid >> 2, cc4 = tid & 3;  // staging coords (rows cr, cr+64)

  v8f c[2][4] = {};

  // prologue: stage k-tile 0 into buffer 0
#pragma unroll
  for (int j = 0; j < 2; ++j) {
    int r = cr + 64 * j;
    *(uint4*)&As[0][r][cc4 * 8] = ((const uint4*)(A  + (size_t)(bM + r) * K))[cc4];
    *(uint4*)&Bs[0][r][cc4 * 8] = ((const uint4*)(Wt + (size_t)(bN + r) * K))[cc4];
  }

  int buf = 0;
  for (int k0 = 0; k0 < K; k0 += 32, buf ^= 1) {
    __syncthreads();
    const bool nxt = (k0 + 32) < K;
    uint4 ra[2], rb[2];
    if (nxt) {  // issue next tile's global loads early (latency hidden by WMMA)
#pragma unroll
      for (int j = 0; j < 2; ++j) {
        int r = cr + 64 * j;
        ra[j] = ((const uint4*)(A  + (size_t)(bM + r) * K + k0 + 32))[cc4];
        rb[j] = ((const uint4*)(Wt + (size_t)(bN + r) * K + k0 + 32))[cc4];
      }
      if (k0 + 64 < K) {  // 2-ahead prefetch (global_prefetch_b8)
        __builtin_prefetch(A  + (size_t)(bM + cr) * K + k0 + 64, 0, 1);
        __builtin_prefetch(Wt + (size_t)(bN + cr) * K + k0 + 64, 0, 1);
      }
    }

    v16us a[2], b[4];
#pragma unroll
    for (int mi = 0; mi < 2; ++mi) {
      int m = wm * 32 + mi * 16 + nl;
#pragma unroll
      for (int i = 0; i < 16; ++i) a[mi][i] = As[buf][m][kA(i, lane)];
    }
#pragma unroll
    for (int ni = 0; ni < 4; ++ni) {
      int n = wn * 64 + ni * 16 + nl;
#pragma unroll
      for (int i = 0; i < 16; ++i) b[ni][i] = Bs[buf][n][kB(i, lane)];
    }
#pragma unroll
    for (int mi = 0; mi < 2; ++mi)
#pragma unroll
      for (int ni = 0; ni < 4; ++ni)
        c[mi][ni] = wmma_bf16(a[mi], b[ni], c[mi][ni]);

    if (nxt) {
#pragma unroll
      for (int j = 0; j < 2; ++j) {
        int r = cr + 64 * j;
        *(uint4*)&As[buf ^ 1][r][cc4 * 8] = ra[j];
        *(uint4*)&Bs[buf ^ 1][r][cc4 * 8] = rb[j];
      }
    }
  }

  const int hi = lane >> 4;
#pragma unroll
  for (int mi = 0; mi < 2; ++mi)
#pragma unroll
    for (int ni = 0; ni < 4; ++ni)
#pragma unroll
      for (int r = 0; r < 8; ++r) {
        int row = bM + wm * 32 + mi * 16 + r + 8 * hi;
        int col = bN + wn * 64 + ni * 16 + nl;
        float v = c[mi][ni][r] + bias[col];
        if (MODE == 0) {
          ((float*)out0)[(size_t)row * N + col] = v;
        } else if (MODE == 1) {
          ((ushort_t*)out0)[(size_t)row * N + col] = f2bf(fmaxf(v, 0.0f));
        } else {  // head layouts: row = b*S+s, col = h*DK+dk
          int bb = row >> 11, s = row & (S_ - 1);
          int h  = col >> 6,  dk = col & (DK_ - 1);
          size_t hidx = (((size_t)(bb * H_ + h) * S_) + s) * DK_ + dk;
          if (MODE == 2) {
            ((float*)out0)[hidx]    = v;
            ((ushort_t*)out1)[hidx] = f2bf(v);
          } else if (MODE == 3) {
            ((ushort_t*)out0)[hidx] = f2bf(v * 0.125f);
          } else {  // MODE 4: V, bf16 shadow transposed to [B,H,DK,S]
            ((float*)out0)[hidx] = v;
            size_t tidx = ((size_t)(bb * H_ + h) * DK_ + dk) * S_ + s;
            ((ushort_t*)out1)[tidx] = f2bf(v);
          }
        }
      }
}

// ---------------------------------------------------------------------------
// Flash attention (causal). Grid (S/128, B*H). 8 waves/block, 16 q-rows/wave.
// Q pre-scaled bf16 [B,H,S,DK]; K bf16 [B,H,S,DK]; V bf16 [B,H,DK,S].
// K staged K-row-major, V staged dim-major -> all frag gathers b128.
// LDS double-buffered, one barrier per key block. Softmax denominator via a
// ones-column WMMA; running max via permlane16.
// ---------------------------------------------------------------------------
__global__ __launch_bounds__(256) void attn_flash_wmma(
    const ushort_t* __restrict__ Qh, const ushort_t* __restrict__ Kh,
    const ushort_t* __restrict__ Vt, ushort_t* __restrict__ concat) {
  __shared__ __align__(16) ushort_t Ks[2][32][72];   // [key][dim]
  __shared__ __align__(16) ushort_t Vs[2][64][40];   // [dim][key]
  __shared__ __align__(16) ushort_t Ps[8][16][34];
  const int tid = threadIdx.x, lane = tid & 31, w = tid >> 5;
  const int bh = blockIdx.y;
  const int bb = bh / H_, h = bh % H_;
  const ushort_t* Qb  = Qh + (size_t)bh * S_ * DK_;
  const ushort_t* Kb  = Kh + (size_t)bh * S_ * DK_;
  const ushort_t* Vbh = Vt + (size_t)bh * DK_ * S_;
  const int qbase = blockIdx.x * 128 + w * 16;
  const int hi = lane >> 4, nl = lane & 15;
  const int kr = tid >> 3, kc = tid & 7;   // K staging: 32 rows x 8 chunks
  const int vd = tid >> 2, vc = tid & 3;   // V staging: 64 dims x 4 chunks

  // Q fragments (bf16, pre-scaled): one-time gather, kept in registers
  v16us aq[2];
  {
    int qm = qbase + nl;
#pragma unroll
    for (int f = 0; f < 2; ++f)
#pragma unroll
      for (int i = 0; i < 16; ++i)
        aq[f][i] = Qb[(size_t)qm * DK_ + f * 32 + kA(i, lane)];
  }

  // Constant B fragment: 32x16 matrix whose column 0 is all ones.
  v16us vones;
#pragma unroll
  for (int i = 0; i < 16; ++i) vones[i] = (nl == 0) ? 0x3F80u : 0u;

  float mrow[8];
  v8f acc[5] = {};  // acc[0..3]: ctx (DK=64), acc[4]: softmax denominator
#pragma unroll
  for (int r = 0; r < 8; ++r) mrow[r] = -3.0e38f;

  // prologue: stage key block 0 into buffer 0
  *(uint4*)&Ks[0][kr][kc * 8] = ((const uint4*)(Kb + (size_t)kr * DK_))[kc];
  *(uint4*)&Vs[0][vd][vc * 8] = ((const uint4*)(Vbh + (size_t)vd * S_))[vc];

  const int nkb = (blockIdx.x + 1) * 4;  // causal: keys [0, qblock_end)
  int buf = 0;
  for (int kb = 0; kb < nkb; ++kb, buf ^= 1) {
    __syncthreads();
    const bool nxt = (kb + 1) < nkb;
    uint4 rk, rv;
    if (nxt) {
      rk = ((const uint4*)(Kb + (size_t)((kb + 1) * 32 + kr) * DK_))[kc];
      rv = ((const uint4*)(Vbh + (size_t)vd * S_ + (kb + 1) * 32))[vc];
    }

    // scores: S(16x32) = Q(16x64) @ K^T
    v8f sc[2];
#pragma unroll
    for (int t = 0; t < 2; ++t) {
      v16us blo, bhi2;
#pragma unroll
      for (int i = 0; i < 16; ++i) {
        blo[i]  = Ks[buf][t * 16 + nl][kB(i, lane)];
        bhi2[i] = Ks[buf][t * 16 + nl][32 + kB(i, lane)];
      }
      v8f z = {};
      sc[t] = wmma_bf16(aq[0], blo, z);
      sc[t] = wmma_bf16(aq[1], bhi2, sc[t]);
    }

    // causal mask + online max update (row reductions via permlane16)
#pragma unroll
    for (int r = 0; r < 8; ++r) {
      int m_local = r + 8 * hi;
      int qidx = qbase + m_local;
      float s0 = (kb * 32 + nl      <= qidx) ? sc[0][r] : -1e9f;
      float s1 = (kb * 32 + 16 + nl <= qidx) ? sc[1][r] : -1e9f;
      float bm = rowmax16(fmaxf(s0, s1));
      float nm = fmaxf(mrow[r], bm);
      float alpha = __expf(mrow[r] - nm);
      float p0 = __expf(s0 - nm), p1 = __expf(s1 - nm);
      mrow[r] = nm;
#pragma unroll
      for (int f = 0; f < 5; ++f) acc[f][r] *= alpha;
      Ps[w][m_local][nl]      = f2bf(p0);  // C-layout -> LDS (bf16)
      Ps[w][m_local][16 + nl] = f2bf(p1);
    }

    // ctx += P(16x32) @ V(32x64) (+ ones column)
    v16us pf;
#pragma unroll
    for (int i = 0; i < 16; ++i) pf[i] = Ps[w][nl][kA(i, lane)];
#pragma unroll
    for (int f = 0; f < 4; ++f) {
      v16us vb;
#pragma unroll
      for (int i = 0; i < 16; ++i) vb[i] = Vs[buf][f * 16 + nl][kB(i, lane)];
      acc[f] = wmma_bf16(pf, vb, acc[f]);
    }
    acc[4] = wmma_bf16(pf, vones, acc[4]);

    if (nxt) {
      *(uint4*)&Ks[buf ^ 1][kr][kc * 8] = rk;
      *(uint4*)&Vs[buf ^ 1][vd][vc * 8] = rv;
    }
  }

  // normalize (broadcast denominator from lane 0 of each 16-lane row) and
  // write ctx in concat [B,S,H*DK] layout as bf16 (feeds the Wo GEMM)
#pragma unroll
  for (int f = 0; f < 4; ++f)
#pragma unroll
    for (int r = 0; r < 8; ++r) {
      int m_local = r + 8 * hi;
      int qidx = qbase + m_local;
      int d = h * DK_ + f * 16 + nl;
      float linv = 1.0f / plane16(acc[4][r], 0, 0);
      concat[((size_t)bb * S_ + qidx) * D_ + d] = f2bf(acc[f][r] * linv);
    }
}

// ---------------------------------------------------------------------------
// out = LayerNorm(a + b) * g + beta   (one row / block); optionally also emit
// a bf16 shadow copy (WB=1) for consumption by a following GEMM.
// ---------------------------------------------------------------------------
template <int WB>
__global__ __launch_bounds__(256) void add_layernorm(
    const float* __restrict__ a, const float* __restrict__ b,
    const float* __restrict__ g, const float* __restrict__ be,
    float* __restrict__ out, ushort_t* __restrict__ outb) {
  __shared__ float s1[256], s2[256];
  const int row = blockIdx.x, tid = threadIdx.x;
  const float* ar = a + (size_t)row * D_;
  const float* br = b + (size_t)row * D_;
  float x[4], sum = 0.f, sq = 0.f;
#pragma unroll
  for (int j = 0; j < 4; ++j) {
    int col = tid + j * 256;
    x[j] = ar[col] + br[col];
    sum += x[j];
    sq  += x[j] * x[j];
  }
  s1[tid] = sum; s2[tid] = sq;
  __syncthreads();
  for (int st = 128; st > 0; st >>= 1) {
    if (tid < st) { s1[tid] += s1[tid + st]; s2[tid] += s2[tid + st]; }
    __syncthreads();
  }
  float mu  = s1[0] * (1.0f / D_);
  float var = s2[0] * (1.0f / D_) - mu * mu;
  float inv = rsqrtf(var + 1e-5f);
  float* orow = out + (size_t)row * D_;
#pragma unroll
  for (int j = 0; j < 4; ++j) {
    int col = tid + j * 256;
    float v = (x[j] - mu) * inv * g[col] + be[col];
    orow[col] = v;
    if (WB) outb[(size_t)row * D_ + col] = f2bf(v);
  }
}

// ---------------------------------------------------------------------------
extern "C" void kernel_launch(void* const* d_in, const int* in_sizes, int n_in,
                              void* d_out, int out_size, void* d_ws, size_t ws_size,
                              hipStream_t stream) {
  (void)in_sizes; (void)n_in; (void)out_size; (void)ws_size;
  const float* query = (const float*)d_in[0];
  const float* key   = (const float*)d_in[1];
  const float* value = (const float*)d_in[2];
  // d_in[3]=batch_size, d_in[4]=sequence_length, d_in[5]=mask (causal handled analytically)
  const float* Wq = (const float*)d_in[6];
  const float* bq = (const float*)d_in[7];
  const float* Wk = (const float*)d_in[8];
  const float* bk = (const float*)d_in[9];
  const float* Wv = (const float*)d_in[10];
  const float* bv = (const float*)d_in[11];
  const float* Wo = (const float*)d_in[12];
  const float* bo = (const float*)d_in[13];
  const float* ln1g = (const float*)d_in[14];
  const float* ln1b = (const float*)d_in[15];
  const float* W1 = (const float*)d_in[16];
  const float* b1 = (const float*)d_in[17];
  const float* W2 = (const float*)d_in[18];
  const float* b2 = (const float*)d_in[19];
  const float* ln2g = (const float*)d_in[20];
  const float* ln2b = (const float*)d_in[21];

  const int NTOK = B_ * S_;                  // 4096
  const size_t TSZ = (size_t)NTOK * D_;      // 4,194,304 elements

  float* out  = (float*)d_out;
  float* x2   = out;               // output 0: [B,S,D]
  float* Kout = out + TSZ;         // output 1: [B,H,S,DK] (f32)
  float* Vout = out + 2 * TSZ;     // output 2: [B,H,S,DK] (f32)

  // workspace layout (bytes)
  char* base = (char*)d_ws;
  const size_t MB = 1u << 20;
  ushort_t* qb     = (ushort_t*)(base + 0 * MB);    // bf16 query   [B,S,D]   8MB
  ushort_t* kbuf   = (ushort_t*)(base + 8 * MB);    // bf16 key                8MB
  ushort_t* vbuf   = (ushort_t*)(base + 16 * MB);   // bf16 value              8MB
  ushort_t* WqT    = (ushort_t*)(base + 24 * MB);   // bf16 W^T                2MB each
  ushort_t* WkT    = (ushort_t*)(base + 26 * MB);
  ushort_t* WvT    = (ushort_t*)(base + 28 * MB);
  ushort_t* WoT    = (ushort_t*)(base + 30 * MB);
  ushort_t* W1T    = (ushort_t*)(base + 32 * MB);   // [DFF,D]  8MB
  ushort_t* W2T    = (ushort_t*)(base + 40 * MB);   // [D,DFF]  8MB
  ushort_t* Qhb    = (ushort_t*)(base + 48 * MB);   // bf16 Q [B,H,S,DK] scaled 8MB
  ushort_t* Kbb    = (ushort_t*)(base + 56 * MB);   // bf16 K [B,H,S,DK]       8MB
  ushort_t* Vtb    = (ushort_t*)(base + 64 * MB);   // bf16 V [B,H,DK,S]       8MB
  ushort_t* ccat   = (ushort_t*)(base + 72 * MB);   // bf16 attention output   8MB
  float*    mha    = (float*)   (base + 80 * MB);   // f32                    16MB
  float*    x1     = (float*)   (base + 96 * MB);   // f32                    16MB
  ushort_t* x1b    = (ushort_t*)(base + 112 * MB);  // bf16 shadow of x1       8MB
  ushort_t* hbufb  = (ushort_t*)(base + 120 * MB);  // bf16 FFN hidden        32MB
  float*    fbuf   = (float*)   (base + 80 * MB);   // reuse mha (dead by then)

  dim3 blk(256);

  // one-time bf16 conversions: activations (row-major) + weights (transposed)
  cast_f32_bf16<<<dim3((int)(TSZ / 2048)), blk, 0, stream>>>(query, qb,   (int)(TSZ / 8));
  cast_f32_bf16<<<dim3((int)(TSZ / 2048)), blk, 0, stream>>>(key,   kbuf, (int)(TSZ / 8));
  cast_f32_bf16<<<dim3((int)(TSZ / 2048)), blk, 0, stream>>>(value, vbuf, (int)(TSZ / 8));
  cast_transpose_bf16<<<dim3(D_ / 64, D_ / 64), blk, 0, stream>>>(Wq, WqT, D_, D_);
  cast_transpose_bf16<<<dim3(D_ / 64, D_ / 64), blk, 0, stream>>>(Wk, WkT, D_, D_);
  cast_transpose_bf16<<<dim3(D_ / 64, D_ / 64), blk, 0, stream>>>(Wv, WvT, D_, D_);
  cast_transpose_bf16<<<dim3(D_ / 64, D_ / 64), blk, 0, stream>>>(Wo, WoT, D_, D_);
  cast_transpose_bf16<<<dim3(DFF_ / 64, D_ / 64), blk, 0, stream>>>(W1, W1T, D_, DFF_);
  cast_transpose_bf16<<<dim3(D_ / 64, DFF_ / 64), blk, 0, stream>>>(W2, W2T, DFF_, D_);

  dim3 gD(D_ / 128, NTOK / 128);    // (8, 32)
  dim3 gF(DFF_ / 128, NTOK / 128);  // (32, 32)

  // QKV projections: Q -> bf16 scaled; K -> f32 + bf16; V -> f32 + bf16^T
  gemm_bf16_wmma<3><<<gD, blk, 0, stream>>>(qb,   WqT, bq, Qhb,  nullptr, NTOK, D_, D_);
  gemm_bf16_wmma<2><<<gD, blk, 0, stream>>>(kbuf, WkT, bk, Kout, Kbb,     NTOK, D_, D_);
  gemm_bf16_wmma<4><<<gD, blk, 0, stream>>>(vbuf, WvT, bv, Vout, Vtb,     NTOK, D_, D_);

  // causal flash attention -> concat [B,S,D] bf16
  attn_flash_wmma<<<dim3(S_ / 128, B_ * H_), blk, 0, stream>>>(Qhb, Kbb, Vtb, ccat);

  // output projection + residual/LN1 (also emits bf16 shadow of x1)
  gemm_bf16_wmma<0><<<gD, blk, 0, stream>>>(ccat, WoT, bo, mha, nullptr, NTOK, D_, D_);
  add_layernorm<1><<<dim3(NTOK), blk, 0, stream>>>(mha, query, ln1g, ln1b, x1, x1b);

  // FFN: ReLU(x1 @ W1 + b1) (bf16) @ W2 + b2, then residual/LN2 -> x2
  gemm_bf16_wmma<1><<<gF, blk, 0, stream>>>(x1b,   W1T, b1, hbufb, nullptr, NTOK, DFF_, D_);
  gemm_bf16_wmma<0><<<gD, blk, 0, stream>>>(hbufb, W2T, b2, fbuf,  nullptr, NTOK, D_, DFF_);
  add_layernorm<0><<<dim3(NTOK), blk, 0, stream>>>(fbuf, x1, ln2g, ln2b, x2, nullptr);
}
